// Attention2_79310866087966
// MI455X (gfx1250) — matrix operand order repaired and checked by reading, not compiled, not run on hardware
//
#include <hip/hip_runtime.h>
#include <hip/hip_bf16.h>
#include <cstdint>

typedef __bf16 bf16;
typedef __attribute__((ext_vector_type(4)))  bf16  v4bf;
typedef __attribute__((ext_vector_type(8)))  bf16  v8bf;
typedef __attribute__((ext_vector_type(16))) bf16  v16bf;
typedef __attribute__((ext_vector_type(8)))  float v8f;

union V16 { v16bf v; v8bf h[2]; };

// ---------------------------------------------------------------------------
// WMMA helpers (CDNA5: D = A[16x32] * B[32x16] + C, bf16 in, f32 accum)
// A operand: lane L holds row (L%16); K chunks at (L/16)*8+{0..7} and 16+(L/16)*8+{0..7}
// B operand: lane L holds column (L%16) (= row of B^T); K chunks at (L/16)*8+{0..7} and 16+(L/16)*8+{0..7}
// C/D: lane L, vgpr v -> element (m = v + 8*(L/16), n = L%16)
// ---------------------------------------------------------------------------
static __device__ __forceinline__ v8f wmma_bf16(v16bf a, v16bf b, v8f c) {
  return __builtin_amdgcn_wmma_f32_16x16x32_bf16(false, a, false, b, (short)0, c,
                                                 false, false);
}

static __device__ __forceinline__ v8f vzero8f() {
  v8f z = {0.f, 0.f, 0.f, 0.f, 0.f, 0.f, 0.f, 0.f};
  return z;
}

static __device__ __forceinline__ v16bf load_a_frag(const bf16* __restrict__ base,
                                                    int ld, int row, int k0, int half) {
  V16 u;
  const bf16* p = base + (size_t)row * ld + k0 + half * 8;
  u.h[0] = *reinterpret_cast<const v8bf*>(p);
  u.h[1] = *reinterpret_cast<const v8bf*>(p + 16);
  return u.v;
}

static __device__ __forceinline__ v16bf load_b_frag(const bf16* __restrict__ base,
                                                    int ld, int row, int k0, int half) {
  V16 u;
  const bf16* p = base + (size_t)row * ld + k0 + half * 8;
  u.h[0] = *reinterpret_cast<const v8bf*>(p);
  u.h[1] = *reinterpret_cast<const v8bf*>(p + 16);
  return u.v;
}

// ---------------------------------------------------------------------------
// CDNA5 async global->LDS copy (ASYNCcnt path). LDS address = low 32 bits of
// the generic pointer (flat->LDS mapping uses addr[31:0]).
// ---------------------------------------------------------------------------
static __device__ __forceinline__ void async_load_b128(uint32_t lds_off,
                                                       const bf16* gptr) {
  asm volatile("global_load_async_to_lds_b128 %0, %1, off"
               :
               : "v"(lds_off), "v"((unsigned long long)(uintptr_t)gptr)
               : "memory");
}
static __device__ __forceinline__ void wait_async_0() {
  asm volatile("s_wait_asynccnt 0x0" ::: "memory");
}
static __device__ __forceinline__ void wait_async_2() {
  asm volatile("s_wait_asynccnt 0x2" ::: "memory");
}

// ---------------------------------------------------------------------------
// fp32 -> bf16 conversion (vector-of-4 per thread)
// ---------------------------------------------------------------------------
__global__ __launch_bounds__(256) void f32_to_bf16_kernel(const float* __restrict__ in,
                                                          bf16* __restrict__ out, int n4) {
  int i = blockIdx.x * 256 + threadIdx.x;
  if (i < n4) {
    float4 f = reinterpret_cast<const float4*>(in)[i];
    v4bf o = {(bf16)f.x, (bf16)f.y, (bf16)f.z, (bf16)f.w};
    *reinterpret_cast<v4bf*>(out + 4 * (size_t)i) = o;
  }
}

// ---------------------------------------------------------------------------
// Core: one wave computes a 64x64 tile of C = A[M,K] * W[N,K]^T (both bf16
// row-major over K), f32 accumulators, 16 WMMA per 32-wide K step.
// ---------------------------------------------------------------------------
static __device__ __forceinline__ void wave_gemm_64x64(const bf16* __restrict__ A, int lda,
                                                       const bf16* __restrict__ W, int ldw,
                                                       int m0, int n0, int K,
                                                       v8f acc[4][4]) {
  const int lane = threadIdx.x & 31;
  const int r = lane & 15;
  const int half = lane >> 4;
#pragma unroll
  for (int i = 0; i < 4; ++i)
#pragma unroll
    for (int j = 0; j < 4; ++j) acc[i][j] = vzero8f();

  for (int k0 = 0; k0 < K; k0 += 32) {
    v16bf a[4], b[4];
#pragma unroll
    for (int i = 0; i < 4; ++i) a[i] = load_a_frag(A, lda, m0 + i * 16 + r, k0, half);
#pragma unroll
    for (int j = 0; j < 4; ++j) b[j] = load_b_frag(W, ldw, n0 + j * 16 + r, k0, half);
#pragma unroll
    for (int i = 0; i < 4; ++i)
#pragma unroll
      for (int j = 0; j < 4; ++j) acc[i][j] = wmma_bf16(a[i], b[j], acc[i][j]);
  }
}

// ---------------------------------------------------------------------------
// QKV projection: [4096,1024] x [3072,1024]^T -> scatter into per-head
//   Q [bh][n][d], K [bh][n][d], Vt [bh][d][n]   (bh = b*16 + h, bf16)
// grid = (3072/128, 4096/256), block = 256 (8 waves as 4x2 of 64x64 tiles)
// ---------------------------------------------------------------------------
__global__ __launch_bounds__(256) void qkv_gemm_kernel(const bf16* __restrict__ xb,
                                                       const bf16* __restrict__ wqkv,
                                                       bf16* __restrict__ qb,
                                                       bf16* __restrict__ kb,
                                                       bf16* __restrict__ vtb) {
  const int wave = threadIdx.x >> 5;
  const int m0 = blockIdx.y * 256 + (wave >> 1) * 64;
  const int n0 = blockIdx.x * 128 + (wave & 1) * 64;
  v8f acc[4][4];
  wave_gemm_64x64(xb, 1024, wqkv, 1024, m0, n0, 1024, acc);

  const int lane = threadIdx.x & 31;
  const int r = lane & 15;
  const int half = lane >> 4;
#pragma unroll
  for (int j = 0; j < 4; ++j) {
    const int gc = n0 + j * 16 + r;          // [0,3072)
    const int which = gc >> 10;              // 0=q 1=k 2=v (uniform per j)
    const int c = gc & 1023;
    const int hh = c >> 6, d = c & 63;
    const int offj_qk = hh * (2048 * 64) + d;
    const int offj_vt = hh * (64 * 2048) + d * 2048;
    bf16* dst = (which == 0) ? qb : ((which == 1) ? kb : vtb);
    const bool isv = (which == 2);
#pragma unroll
    for (int i = 0; i < 4; ++i)
#pragma unroll
      for (int v = 0; v < 8; ++v) {
        const int gr = m0 + i * 16 + v + 8 * half;   // [0,4096)
        const int b = gr >> 11, nq = gr & 2047;
        const int base_b = b * (16 * 2048 * 64);
        const int off = isv ? (base_b + offj_vt + nq)
                            : (base_b + offj_qk + nq * 64);
        dst[off] = (bf16)acc[i][j][v];
      }
  }
}

// ---------------------------------------------------------------------------
// Flash attention with double-buffered async global->LDS staging of the K/V
// tiles shared by all 8 waves of a block.
// grid = (2048/128, 32); block = 256; one wave owns 16 queries.
// S^T = K*Q^T so softmax stats live per-lane (query = L%16); P bounced through
// a per-wave LDS slab to re-shape D-layout -> A-layout; H += P*V with Vt.
// ---------------------------------------------------------------------------
#define KB_STRIDE 72  // bf16/row for K tile  (144B: 16B aligned, bank-spread)
#define VT_STRIDE 40  // bf16/row for Vt tile (80B)
#define P_STRIDE 40   // bf16/row for P slab

__global__ __launch_bounds__(256) void attn_kernel(const bf16* __restrict__ qb,
                                                   const bf16* __restrict__ kb,
                                                   const bf16* __restrict__ vtb,
                                                   bf16* __restrict__ hb) {
  __shared__ __align__(16) bf16 Kbuf[2][32 * KB_STRIDE];   // 32 keys x 64 d
  __shared__ __align__(16) bf16 Vtbuf[2][64 * VT_STRIDE];  // 64 d x 32 keys
  __shared__ __align__(16) bf16 Pbuf[8][16 * P_STRIDE];

  const int tid = threadIdx.x;
  const int wave = tid >> 5;
  const int lane = tid & 31;
  const int r = lane & 15;
  const int half = lane >> 4;
  const int bh = blockIdx.y;                 // b*16 + h
  const int q0 = blockIdx.x * 128 + wave * 16;

  const bf16* Q  = qb  + (size_t)bh * 2048 * 64;
  const bf16* Kp = kb  + (size_t)bh * 2048 * 64;
  const bf16* Vt = vtb + (size_t)bh * 64 * 2048;
  bf16* myP = Pbuf[wave];

  // cooperative tile-copy assignments (256 threads, 1 b128 chunk each per buf)
  const int krow = tid >> 3, kcol = (tid & 7) * 8;  // K: 32 rows x 8 chunks of 8 d
  const int vrow = tid >> 2, vcol = (tid & 3) * 8;  // Vt: 64 rows x 4 chunks of 8 keys

  // Q as B operand (B^T = Q[query, d] row-major): d-chunks [0,32) and [32,64)
  v16bf qf0 = load_b_frag(Q, 64, q0 + r, 0, half);
  v16bf qf1 = load_b_frag(Q, 64, q0 + r, 32, half);

  float mrun = -3.0e38f, lrun = 0.0f;
  v8f hacc[4];
#pragma unroll
  for (int t = 0; t < 4; ++t) hacc[t] = vzero8f();

  const float scale = 0.125f;  // D^-0.5, D = 64

  // prologue: stage tile 0 into buffer 0
  async_load_b128((uint32_t)(uintptr_t)&Kbuf[0][krow * KB_STRIDE + kcol],
                  Kp + (size_t)krow * 64 + kcol);
  async_load_b128((uint32_t)(uintptr_t)&Vtbuf[0][vrow * VT_STRIDE + vcol],
                  Vt + (size_t)vrow * 2048 + vcol);

  for (int kt = 0; kt < 64; ++kt) {
    const int cur = kt & 1;
    if (kt + 1 < 64) {  // stream next tile into the other buffer
      const int nb = cur ^ 1;
      const int nk = (kt + 1) * 32;
      async_load_b128((uint32_t)(uintptr_t)&Kbuf[nb][krow * KB_STRIDE + kcol],
                      Kp + (size_t)(nk + krow) * 64 + kcol);
      async_load_b128((uint32_t)(uintptr_t)&Vtbuf[nb][vrow * VT_STRIDE + vcol],
                      Vt + (size_t)vrow * 2048 + nk + vcol);
      wait_async_2();  // tile kt's 2 ops done; next tile's 2 still in flight
    } else {
      wait_async_0();
    }
    __syncthreads();   // all waves' copies for tile kt visible

    // S^T tiles (keys x queries) from LDS K tile + register Q frags
    v8f s[2];
#pragma unroll
    for (int sub = 0; sub < 2; ++sub) {
      v16bf ka0 = load_a_frag(&Kbuf[cur][0], KB_STRIDE, sub * 16 + r, 0, half);
      v16bf ka1 = load_a_frag(&Kbuf[cur][0], KB_STRIDE, sub * 16 + r, 32, half);
      v8f a = vzero8f();
      a = wmma_bf16(ka0, qf0, a);
      a = wmma_bf16(ka1, qf1, a);
      s[sub] = a;
    }

    // --- online softmax (per-lane stats for query r; halves merged) ---
    float sc[2][8];
    float tm = -3.0e38f;
#pragma unroll
    for (int sub = 0; sub < 2; ++sub)
#pragma unroll
      for (int v = 0; v < 8; ++v) {
        sc[sub][v] = s[sub][v] * scale;
        tm = fmaxf(tm, sc[sub][v]);
      }
    tm = fmaxf(tm, __shfl_xor(tm, 16, 32));
    float nm = fmaxf(mrun, tm);
    float alpha = __expf(mrun - nm);
    mrun = nm;

    float ts = 0.0f;
    v8bf p[2];
#pragma unroll
    for (int sub = 0; sub < 2; ++sub)
#pragma unroll
      for (int v = 0; v < 8; ++v) {
        float e = __expf(sc[sub][v] - nm);
        ts += e;
        p[sub][v] = (bf16)e;
      }
    ts += __shfl_xor(ts, 16, 32);
    lrun = lrun * alpha + ts;

    // rescale H accumulators: row m = v + 8*half needs alpha of query m
    float av[8];
#pragma unroll
    for (int v = 0; v < 8; ++v) av[v] = __shfl(alpha, v + 8 * half, 32);
#pragma unroll
    for (int t = 0; t < 4; ++t)
#pragma unroll
      for (int v = 0; v < 8; ++v) hacc[t][v] *= av[v];

    // P^T tile -> per-wave LDS slab as P[query][key_local] (contiguous in v)
    *reinterpret_cast<v8bf*>(&myP[r * P_STRIDE + 8 * half])      = p[0];
    *reinterpret_cast<v8bf*>(&myP[r * P_STRIDE + 16 + 8 * half]) = p[1];

    // re-read P as WMMA A operand (16 queries x 32 keys)
    V16 u;
    u.h[0] = *reinterpret_cast<const v8bf*>(&myP[r * P_STRIDE + half * 8]);
    u.h[1] = *reinterpret_cast<const v8bf*>(&myP[r * P_STRIDE + 16 + half * 8]);
    v16bf pf = u.v;

    // H[queries, d] += P * V ; B^T = Vt tile rows (d x keys) from LDS
#pragma unroll
    for (int t = 0; t < 4; ++t) {
      v16bf vf = load_b_frag(&Vtbuf[cur][0], VT_STRIDE, t * 16 + r, 0, half);
      hacc[t] = wmma_bf16(pf, vf, hacc[t]);
    }

    __syncthreads();   // everyone done reading buffer `cur` before reuse
  }

  // finalize: divide by row sums, store bf16 h[b, n, c]
  float rl = 1.0f / lrun;
  float rv[8];
#pragma unroll
  for (int v = 0; v < 8; ++v) rv[v] = __shfl(rl, v + 8 * half, 32);
#pragma unroll
  for (int t = 0; t < 4; ++t)
#pragma unroll
    for (int v = 0; v < 8; ++v) {
      int qg = q0 + v + 8 * half;
      int c = (bh & 15) * 64 + t * 16 + r;
      hb[((size_t)((bh >> 4) * 2048 + qg)) * 1024 + c] = (bf16)(hacc[t][v] * rv[v]);
    }
}

// ---------------------------------------------------------------------------
// Output projection: out[4096,1024] = h * proj_w^T + proj_b  (f32 out)
// ---------------------------------------------------------------------------
__global__ __launch_bounds__(256) void proj_gemm_kernel(const bf16* __restrict__ hb,
                                                        const bf16* __restrict__ wp,
                                                        const float* __restrict__ bias,
                                                        float* __restrict__ out) {
  const int wave = threadIdx.x >> 5;
  const int m0 = blockIdx.y * 256 + (wave >> 1) * 64;
  const int n0 = blockIdx.x * 128 + (wave & 1) * 64;
  v8f acc[4][4];
  wave_gemm_64x64(hb, 1024, wp, 1024, m0, n0, 1024, acc);

  const int lane = threadIdx.x & 31;
  const int r = lane & 15;
  const int half = lane >> 4;
#pragma unroll
  for (int j = 0; j < 4; ++j) {
    const int gc = n0 + j * 16 + r;
    const float bj = bias[gc];
#pragma unroll
    for (int i = 0; i < 4; ++i)
#pragma unroll
      for (int v = 0; v < 8; ++v) {
        const int gr = m0 + i * 16 + v + 8 * half;
        out[gr * 1024 + gc] = acc[i][j][v] + bj;
      }
  }
}

// ---------------------------------------------------------------------------
// Host launch
// ---------------------------------------------------------------------------
extern "C" void kernel_launch(void* const* d_in, const int* in_sizes, int n_in,
                              void* d_out, int out_size, void* d_ws, size_t ws_size,
                              hipStream_t stream) {
  (void)in_sizes; (void)n_in; (void)out_size; (void)ws_size;
  const float* x      = (const float*)d_in[0];  // [2,2048,1024]
  const float* qkv_w  = (const float*)d_in[1];  // [3072,1024]
  const float* proj_w = (const float*)d_in[2];  // [1024,1024]
  const float* proj_b = (const float*)d_in[3];  // [1024]

  const size_t XB = 4096u * 1024u;
  const size_t WQ = 3072u * 1024u;
  const size_t WP = 1024u * 1024u;
  const size_t QK = 2u * 16u * 2048u * 64u;

  char* ws = (char*)d_ws;
  bf16* xb    = (bf16*)ws;  ws += XB * sizeof(bf16);  // reused as h-buffer
  bf16* wqkvb = (bf16*)ws;  ws += WQ * sizeof(bf16);
  bf16* wprojb= (bf16*)ws;  ws += WP * sizeof(bf16);
  bf16* qb    = (bf16*)ws;  ws += QK * sizeof(bf16);
  bf16* kb    = (bf16*)ws;  ws += QK * sizeof(bf16);
  bf16* vtb   = (bf16*)ws;  ws += QK * sizeof(bf16);
  bf16* hbuf  = xb;  // x no longer needed after the QKV GEMM

  f32_to_bf16_kernel<<<(int)(XB / 1024), 256, 0, stream>>>(x, xb, (int)(XB / 4));
  f32_to_bf16_kernel<<<(int)(WQ / 1024), 256, 0, stream>>>(qkv_w, wqkvb, (int)(WQ / 4));
  f32_to_bf16_kernel<<<(int)(WP / 1024), 256, 0, stream>>>(proj_w, wprojb, (int)(WP / 4));

  qkv_gemm_kernel<<<dim3(24, 16), 256, 0, stream>>>(xb, wqkvb, qb, kb, vtb);
  attn_kernel<<<dim3(16, 32), 256, 0, stream>>>(qb, kb, vtb, hbuf);
  proj_gemm_kernel<<<dim3(8, 16), 256, 0, stream>>>(hbuf, wprojb, proj_b, (float*)d_out);
}
